// SwinTransformerV2Block_89687507075915
// MI455X (gfx1250) — matrix operand order repaired and checked by reading, not compiled, not run on hardware
//
#include <hip/hip_runtime.h>
#include <math.h>

// SwinTransformerV2 block for MI455X (gfx1250, wave32, WMMA 16x16x32 f16).
// All GEMMs (qkv, attn QK^T, attn PV, proj, fc1, fc2) use
// v_wmma_f32_16x16x32_f16 (f16 inputs, f32 accumulation).
// GEMM tiles are software-pipelined through double-buffered LDS; tile copies
// use GLOBAL_LOAD_ASYNC_TO_LDS_B128 (ASYNCcnt) when the toolchain exposes the
// builtin, falling back to global_load_b128 + ds_store_b128 otherwise.
// Attention materializes the full 256x256 score tile per (window,head) in
// LDS (147 KB) -- enabled by CDNA5's 320 KB LDS per WGP.
// Requires ws_size >= ~350 MB.

typedef _Float16 f16;
typedef __attribute__((ext_vector_type(8)))  _Float16 v8h;
typedef __attribute__((ext_vector_type(16))) _Float16 v16h;
typedef __attribute__((ext_vector_type(8)))  float    v8f;
typedef __attribute__((ext_vector_type(4)))  int      v4i;

// ------------------------------------------------------- async-to-LDS path
// Probe-discovered signature (from clang diagnostic):
//   __builtin_amdgcn_global_load_async_to_lds_b128(
//       int4 __device__* src, int4 __shared__* dst, imm int off, imm int cpol)
#if defined(__HIP_DEVICE_COMPILE__) && __has_builtin(__builtin_amdgcn_global_load_async_to_lds_b128)
#define HAVE_ASYNC_LDS 1
typedef __attribute__((address_space(1))) v4i g_v4i;
typedef __attribute__((address_space(3))) v4i l_v4i;
// generic->AS pointer via integer cast: generic LDS pointers keep the LDS
// offset in the low 32 bits; generic global pointers equal the AS1 address.
#define ASYNC_COPY_B128(gsrc, ldst)                                          \
  __builtin_amdgcn_global_load_async_to_lds_b128(                            \
      (g_v4i*)(uintptr_t)(gsrc), (l_v4i*)(uintptr_t)(ldst), 0, 0)
#if __has_builtin(__builtin_amdgcn_s_wait_asynccnt)
#define ASYNC_WAIT() __builtin_amdgcn_s_wait_asynccnt(0)
#else
#define ASYNC_WAIT() asm volatile("s_wait_asynccnt 0" ::: "memory")
#endif
#else
#define HAVE_ASYNC_LDS 0
#define ASYNC_WAIT()
#endif

// ---------------------------------------------------------------- helpers

static __device__ inline v8f vzero8() {
  v8f z;
#pragma unroll
  for (int i = 0; i < 8; ++i) z[i] = 0.0f;
  return z;
}

static __device__ inline v16h mkfrag(v8h a, v8h b) {
  v16h r;
#pragma unroll
  for (int i = 0; i < 8; ++i) { r[i] = a[i]; r[i + 8] = b[i]; }
  return r;
}

// A-fragment (16x32 f16): lane L holds row m=L%16, half-group h=L/16.
// Element i<8 -> k = 8h+i ; i>=8 -> k = 16+8h+(i-8): two contiguous 16B chunks.
static __device__ inline v16h afrag_from(const f16* row, int hgrp) {
  v8h c0 = *(const v8h*)(row + 8 * hgrp);
  v8h c1 = *(const v8h*)(row + 16 + 8 * hgrp);
  return mkfrag(c0, c1);
}

// B-fragment (32x16 f16): lane L holds column n=L%16, h=L/16; k order as A.
static __device__ inline v16h bfrag_from(const f16* p, int pitch, int hgrp) {
  v16h r;
#pragma unroll
  for (int i = 0; i < 8; ++i) {
    r[i]     = p[(8 * hgrp + i) * pitch];
    r[i + 8] = p[(16 + 8 * hgrp + i) * pitch];
  }
  return r;
}

static __device__ inline v8f wmma16x16x32(v16h a, v16h b, v8f c) {
  return __builtin_amdgcn_wmma_f32_16x16x32_f16(false, a, false, b,
                                                (short)0, c, false, false);
}

static __device__ inline int wregion(int wh, int ww, int t) {
  int gh = wh * 16 + (t >> 4), gw = ww * 16 + (t & 15);
  int rh = (gh < 48) ? 0 : ((gh < 56) ? 1 : 2);
  int rw = (gw < 48) ? 0 : ((gw < 56) ? 1 : 2);
  return rh * 3 + rw;
}

// ------------------------------------------------------------ small kernels

// fp32 (N,K) row-major weight -> fp16 (K,N) so GEMM B-tiles are n-contiguous.
__global__ __launch_bounds__(256) void packw_kernel(const float* __restrict__ Wf,
                                                    f16* __restrict__ Wt,
                                                    int N, int K) {
  size_t tid = (size_t)blockIdx.x * 256 + threadIdx.x;
  if (tid >= (size_t)N * K) return;
  int n = (int)(tid % N);
  size_t k = tid / N;
  Wt[k * N + n] = (f16)Wf[(size_t)n * K + k];
}

__global__ __launch_bounds__(256) void build_qkv_bias(const float* __restrict__ qb,
                                                      const float* __restrict__ vb,
                                                      float* __restrict__ o) {
  int i = blockIdx.x * 256 + threadIdx.x;   // 1536
  o[i] = (i < 512) ? qb[i] : ((i < 1024) ? 0.0f : vb[i - 1024]);
}

// roll(-8,-8) + window partition + f16 convert. token=(((b*4+wh)*4+ww)*256+i*16+j)
__global__ __launch_bounds__(256) void winpart_kernel(const float* __restrict__ x,
                                                      f16* __restrict__ xw) {
  size_t tid = (size_t)blockIdx.x * 256 + threadIdx.x;  // 32768*512
  int c = (int)(tid & 511);
  size_t tok = tid >> 9;
  int ij = (int)(tok & 255);
  int wwi = (int)((tok >> 8) & 3), whi = (int)((tok >> 10) & 3), b = (int)(tok >> 12);
  int i = ij >> 4, j = ij & 15;
  int sr = (whi * 16 + i + 8) & 63, sc = (wwi * 16 + j + 8) & 63;
  xw[tid] = (f16)x[(((size_t)b * 64 + sr) * 64 + sc) * 512 + c];
}

static __device__ inline float relcoord(int v) {
  float t = (float)(v - 15) * (8.0f / 15.0f);
  float a = log2f(fabsf(t) + 1.0f) * (1.0f / 3.0f);   // /log2(8)
  return (t < 0.0f) ? -a : a;
}

// CPB MLP: table row r -> 16*sigmoid(relu(x@W1^T+b1)@W2^T). One WG per row.
__global__ __launch_bounds__(512) void cpb_table_kernel(const float* __restrict__ w1,
                                                        const float* __restrict__ b1,
                                                        const float* __restrict__ w2,
                                                        float* __restrict__ tbl) {
  __shared__ float acc[16];
  int r = blockIdx.x, u = threadIdx.x;
  if (u < 16) acc[u] = 0.0f;
  __syncthreads();
  float x0 = relcoord(r / 31), x1 = relcoord(r % 31);
  float hu = fmaxf(0.0f, w1[2 * u] * x0 + w1[2 * u + 1] * x1 + b1[u]);
#pragma unroll
  for (int h = 0; h < 16; ++h) atomicAdd(&acc[h], hu * w2[h * 512 + u]);
  __syncthreads();
  if (u < 16) tbl[r * 16 + u] = 16.0f / (1.0f + expf(-acc[u]));
}

// rpb[h][n][m] = tbl[idx(n,m)][h]
__global__ __launch_bounds__(256) void rpb_scatter_kernel(const float* __restrict__ tbl,
                                                          float* __restrict__ rpb) {
  int tid = blockIdx.x * 256 + threadIdx.x;  // 65536
  int n = tid >> 8, m = tid & 255;
  int d0 = (n >> 4) - (m >> 4) + 15;
  int d1 = (n & 15) - (m & 15) + 15;
  int idx = d0 * 31 + d1;
#pragma unroll
  for (int h = 0; h < 16; ++h) rpb[(h << 16) + tid] = tbl[idx * 16 + h];
}

// -------------------------------------------------------------- WMMA GEMM
// C[M,N] = A(f16 MxK) @ Bp(f16 KxN) + bias ; epi: 0=f16, 1=f16 gelu, 2=f32
// Block tile 128x128, BK=32, 256 threads = 8 waves, each wave 32x64 (2x4 wmma).
// Double-buffered LDS: copy of tile k+1 (async-to-LDS when available)
// overlaps WMMA compute of tile k; one barrier per K-step.
__global__ __launch_bounds__(256) void gemm_f16(const f16* __restrict__ A,
                                                const f16* __restrict__ Bp,
                                                const float* __restrict__ bias,
                                                void* __restrict__ Cout,
                                                int M, int N, int K, int epi) {
  __shared__ f16 Ah[2][128][40];    // pitch 80B (16B multiple)
  __shared__ f16 Bh[2][32][136];    // pitch 272B
  const int tid = threadIdx.x;
  const int lane = tid & 31, wid = tid >> 5;
  const int wm = wid & 3, wn = wid >> 2;
  const int m0 = blockIdx.y * 128, n0 = blockIdx.x * 128;

  v8f acc[2][4];
#pragma unroll
  for (int i = 0; i < 2; ++i)
#pragma unroll
    for (int j = 0; j < 4; ++j) acc[i][j] = vzero8();

  const int ar = tid >> 1, aseg = (tid & 1) * 16;
  const int br = tid >> 3, bseg = (tid & 7) * 16;
  const int hgrp = lane >> 4;
  const int nk = K >> 5;

  auto copy_tile = [&](int k0, int buf) {
    const f16* as = A + (size_t)(m0 + ar) * K + k0 + aseg;
    const f16* bs = Bp + (size_t)(k0 + br) * N + n0 + bseg;
#if HAVE_ASYNC_LDS
    ASYNC_COPY_B128(as,     &Ah[buf][ar][aseg]);
    ASYNC_COPY_B128(as + 8, &Ah[buf][ar][aseg + 8]);
    ASYNC_COPY_B128(bs,     &Bh[buf][br][bseg]);
    ASYNC_COPY_B128(bs + 8, &Bh[buf][br][bseg + 8]);
#else
    const v8h* av = (const v8h*)as;
    *(v8h*)&Ah[buf][ar][aseg]     = av[0];
    *(v8h*)&Ah[buf][ar][aseg + 8] = av[1];
    const v8h* bv = (const v8h*)bs;
    *(v8h*)&Bh[buf][br][bseg]     = bv[0];
    *(v8h*)&Bh[buf][br][bseg + 8] = bv[1];
#endif
  };

  copy_tile(0, 0);
  ASYNC_WAIT();
  __syncthreads();

  for (int kk = 0; kk < nk; ++kk) {
    const int cur = kk & 1;
    if (kk + 1 < nk) copy_tile((kk + 1) << 5, cur ^ 1);

    v16h af[2];
#pragma unroll
    for (int mt = 0; mt < 2; ++mt)
      af[mt] = afrag_from(&Ah[cur][wm * 32 + mt * 16 + (lane & 15)][0], hgrp);
#pragma unroll
    for (int nt = 0; nt < 4; ++nt) {
      v16h bf = bfrag_from(&Bh[cur][0][wn * 64 + nt * 16 + (lane & 15)], 136, hgrp);
      acc[0][nt] = wmma16x16x32(af[0], bf, acc[0][nt]);
      acc[1][nt] = wmma16x16x32(af[1], bf, acc[1][nt]);
    }
    ASYNC_WAIT();        // tile k+1 landed in LDS
    __syncthreads();     // also: all waves done reading buf cur
  }

#pragma unroll
  for (int mt = 0; mt < 2; ++mt)
#pragma unroll
    for (int nt = 0; nt < 4; ++nt) {
      int gn = n0 + wn * 64 + nt * 16 + (lane & 15);
      float bv = bias ? bias[gn] : 0.0f;
#pragma unroll
      for (int r = 0; r < 8; ++r) {
        int gm = m0 + wm * 32 + mt * 16 + r + 8 * (lane >> 4);
        float v = acc[mt][nt][r] + bv;
        if (epi == 1) v = 0.5f * v * (1.0f + erff(v * 0.70710678118654752f));
        if (epi == 2) ((float*)Cout)[(size_t)gm * N + gn] = v;
        else          ((f16*)Cout)[(size_t)gm * N + gn] = (f16)v;
      }
    }
}

// ---------------------------------------------------- window attention WG
// One workgroup per (window bn, head). 256 threads = 8 waves.
// LDS: scores 256x256 f16 (128KB, reused in place as probabilities)
//      + 16KB union {K^T f16[32][256], V f16[256][32]}  => 147456 B.
__global__ __launch_bounds__(256) void attn_kernel(const f16* __restrict__ qkv,
                                                   const float* __restrict__ logit_scale,
                                                   const float* __restrict__ rpb,
                                                   f16* __restrict__ aout) {
  __shared__ f16 Sh[256][256];
  __shared__ f16 KV[32][256];
  const int bn = blockIdx.x, head = blockIdx.y;
  const int tid = threadIdx.x;
  const int lane = tid & 31, wid = tid >> 5;
  const int widx = bn & 15;
  const int wh = widx >> 2, ww = widx & 3;
  const size_t tokbase = (size_t)bn * 256;
  const int hgrp = lane >> 4;

  // Phase 0: normalize K rows (cosine attention) into KV as K^T[d][key].
  {
    const f16* kr = qkv + (tokbase + tid) * 1536 + 512 + head * 32;
    float kv[32]; float ss = 0.0f;
#pragma unroll
    for (int d = 0; d < 32; ++d) { kv[d] = (float)kr[d]; ss += kv[d] * kv[d]; }
    float sc = 1.0f / fmaxf(sqrtf(ss), 1e-12f);
#pragma unroll
    for (int d = 0; d < 32; ++d) KV[d][tid] = (f16)(kv[d] * sc);
  }
  __syncthreads();

  // Phase 1: S = Qn @ Kn^T. Each wave owns 32 query rows. Q normalized in
  // registers: lanes L and L^16 hold complementary halves of the same row.
  {
    v16h aq[2];
#pragma unroll
    for (int mt = 0; mt < 2; ++mt) {
      int row = wid * 32 + mt * 16 + (lane & 15);
      const f16* qr = qkv + (tokbase + row) * 1536 + head * 32;
      v8h c0 = *(const v8h*)(qr + 8 * hgrp);
      v8h c1 = *(const v8h*)(qr + 16 + 8 * hgrp);
      float ss = 0.0f;
#pragma unroll
      for (int i = 0; i < 8; ++i) {
        float a = (float)c0[i], b = (float)c1[i];
        ss += a * a + b * b;
      }
      ss += __shfl_xor(ss, 16, 32);   // wave32 cross-half reduce
      float sc = 1.0f / fmaxf(sqrtf(ss), 1e-12f);
      v16h f;
#pragma unroll
      for (int i = 0; i < 8; ++i) {
        f[i]     = (f16)((float)c0[i] * sc);
        f[i + 8] = (f16)((float)c1[i] * sc);
      }
      aq[mt] = f;
    }
#pragma unroll 4
    for (int nt = 0; nt < 16; ++nt) {
      v16h bf = bfrag_from(&KV[0][nt * 16 + (lane & 15)], 256, hgrp);
      v8f s0 = wmma16x16x32(aq[0], bf, vzero8());
      v8f s1 = wmma16x16x32(aq[1], bf, vzero8());
#pragma unroll
      for (int r = 0; r < 8; ++r) {
        Sh[wid * 32 +      r + 8 * hgrp][nt * 16 + (lane & 15)] = (f16)s0[r];
        Sh[wid * 32 + 16 + r + 8 * hgrp][nt * 16 + (lane & 15)] = (f16)s1[r];
      }
    }
  }
  __syncthreads();

  // Phase 2: async-load V into KV union (Kt dead) overlapped with in-place
  // softmax of row `tid` (softmax touches only Sh).
  {
    f16 (*Vh)[32] = (f16(*)[32])KV;
    const f16* vr = qkv + (tokbase + tid) * 1536 + 1024 + head * 32;
#if HAVE_ASYNC_LDS
    ASYNC_COPY_B128(vr,      &Vh[tid][0]);
    ASYNC_COPY_B128(vr + 8,  &Vh[tid][8]);
    ASYNC_COPY_B128(vr + 16, &Vh[tid][16]);
    ASYNC_COPY_B128(vr + 24, &Vh[tid][24]);
#else
#pragma unroll
    for (int d = 0; d < 32; ++d) Vh[tid][d] = vr[d];
#endif

    float sc = expf(fminf(logit_scale[head], 4.6051701859880914f)); // log(100)
    int rg_t = wregion(wh, ww, tid);
    const float* rp = rpb + ((size_t)head << 16) + (size_t)tid * 256;
    float mx = -1e30f;
    for (int j = 0; j < 256; ++j) {
      float v = (float)Sh[tid][j] * sc + rp[j] +
                ((rg_t == wregion(wh, ww, j)) ? 0.0f : -100.0f);
      mx = fmaxf(mx, v);
    }
    float sum = 0.0f;
    for (int j = 0; j < 256; ++j) {
      float v = (float)Sh[tid][j] * sc + rp[j] +
                ((rg_t == wregion(wh, ww, j)) ? 0.0f : -100.0f);
      sum += expf(v - mx);
    }
    float inv = 1.0f / sum;
    for (int j = 0; j < 256; ++j) {
      float v = (float)Sh[tid][j] * sc + rp[j] +
                ((rg_t == wregion(wh, ww, j)) ? 0.0f : -100.0f);
      Sh[tid][j] = (f16)(expf(v - mx) * inv);    // in place: own row only
    }
  }
  ASYNC_WAIT();
  __syncthreads();

  // Phase 3: O = P @ V, 256-key loop in chunks of 32.
  {
    f16 (*Vh)[32] = (f16(*)[32])KV;
    v8f o[2][2];
#pragma unroll
    for (int i = 0; i < 2; ++i)
#pragma unroll
      for (int j = 0; j < 2; ++j) o[i][j] = vzero8();

    for (int kt = 0; kt < 8; ++kt) {
      v16h ap[2];
#pragma unroll
      for (int mt = 0; mt < 2; ++mt)
        ap[mt] = afrag_from(&Sh[wid * 32 + mt * 16 + (lane & 15)][kt * 32], hgrp);
#pragma unroll
      for (int nt = 0; nt < 2; ++nt) {
        v16h bf = bfrag_from(&Vh[kt * 32][nt * 16 + (lane & 15)], 32, hgrp);
        o[0][nt] = wmma16x16x32(ap[0], bf, o[0][nt]);
        o[1][nt] = wmma16x16x32(ap[1], bf, o[1][nt]);
      }
    }
#pragma unroll
    for (int mt = 0; mt < 2; ++mt)
#pragma unroll
      for (int nt = 0; nt < 2; ++nt)
#pragma unroll
        for (int r = 0; r < 8; ++r) {
          int row = wid * 32 + mt * 16 + r + 8 * hgrp;
          int col = head * 32 + nt * 16 + (lane & 15);
          aout[(tokbase + row) * 512 + col] = (f16)o[mt][nt][r];
        }
  }
}

// -------------------------------------------------- fused LN + residual
// out32[p] = base[p] + LN(y[row(p)]) * g + b ; optional f16 copy for next GEMM.
// remap=1: p is pixel (b,h,w); row = window token after inverse roll(+8).
__global__ __launch_bounds__(256) void ln_residual(const float* __restrict__ base,
                                                   const float* __restrict__ y,
                                                   const float* __restrict__ g,
                                                   const float* __restrict__ bt,
                                                   float* __restrict__ out32,
                                                   f16* __restrict__ out16,
                                                   int remap) {
  __shared__ float r1[256], r2[256];
  int p = blockIdx.x, t = threadIdx.x;
  int row = p;
  if (remap) {
    int b = p >> 12, hh = (p >> 6) & 63, wp = p & 63;
    int hs = (hh + 56) & 63, ws = (wp + 56) & 63;     // (h-8) mod 64
    row = (((b * 4 + (hs >> 4)) * 4 + (ws >> 4)) * 256) + (hs & 15) * 16 + (ws & 15);
  }
  const float* yr = y + (size_t)row * 512;
  float v0 = yr[t], v1 = yr[t + 256];
  r1[t] = v0 + v1;
  r2[t] = v0 * v0 + v1 * v1;
  __syncthreads();
  for (int s = 128; s > 0; s >>= 1) {
    if (t < s) { r1[t] += r1[t + s]; r2[t] += r2[t + s]; }
    __syncthreads();
  }
  float mean = r1[0] * (1.0f / 512.0f);
  float var  = r2[0] * (1.0f / 512.0f) - mean * mean;
  float rstd = rsqrtf(var + 1e-5f);
  const float* br = base + (size_t)p * 512;
  float o0 = br[t]       + (v0 - mean) * rstd * g[t]       + bt[t];
  float o1 = br[t + 256] + (v1 - mean) * rstd * g[t + 256] + bt[t + 256];
  out32[(size_t)p * 512 + t]       = o0;
  out32[(size_t)p * 512 + t + 256] = o1;
  if (out16) {
    out16[(size_t)p * 512 + t]       = (f16)o0;
    out16[(size_t)p * 512 + t + 256] = (f16)o1;
  }
}

// ------------------------------------------------------------------ launch

extern "C" void kernel_launch(void* const* d_in, const int* in_sizes, int n_in,
                              void* d_out, int out_size, void* d_ws, size_t ws_size,
                              hipStream_t stream) {
  (void)in_sizes; (void)n_in; (void)out_size; (void)ws_size;
  const float* x      = (const float*)d_in[0];
  const float* qkv_w  = (const float*)d_in[1];
  const float* q_bias = (const float*)d_in[2];
  const float* v_bias = (const float*)d_in[3];
  const float* ls     = (const float*)d_in[4];
  const float* cpb_w1 = (const float*)d_in[5];
  const float* cpb_b1 = (const float*)d_in[6];
  const float* cpb_w2 = (const float*)d_in[7];
  const float* proj_w = (const float*)d_in[8];
  const float* proj_b = (const float*)d_in[9];
  const float* n1g    = (const float*)d_in[10];
  const float* n1b    = (const float*)d_in[11];
  const float* n2g    = (const float*)d_in[12];
  const float* n2b    = (const float*)d_in[13];
  const float* fc1_w  = (const float*)d_in[14];
  const float* fc1_b  = (const float*)d_in[15];
  const float* fc2_w  = (const float*)d_in[16];
  const float* fc2_b  = (const float*)d_in[17];
  float* out = (float*)d_out;

  char* w = (char*)d_ws;
  size_t off = 0;
  auto take = [&](size_t bytes) -> void* {
    void* p = w + off;
    off = (off + bytes + 255) & ~(size_t)255;
    return p;
  };

  f16*   wqkv = (f16*)take(512ull * 1536 * 2);
  f16*   wprj = (f16*)take(512ull * 512 * 2);
  f16*   wfc1 = (f16*)take(512ull * 2048 * 2);
  f16*   wfc2 = (f16*)take(2048ull * 512 * 2);
  float* qkvb = (float*)take(1536ull * 4);
  float* tbl  = (float*)take(961ull * 16 * 4);
  float* rpb  = (float*)take(16ull * 256 * 256 * 4);
  f16*   xw   = (f16*)take(32768ull * 512 * 2);   // reused as attn output
  f16*   qkvh = (f16*)take(32768ull * 2048 * 2);  // qkv(1536 cols) then hid(2048)
  float* prj  = (float*)take(32768ull * 512 * 4); // reused as mlp output
  float* x1   = (float*)take(32768ull * 512 * 4);
  f16*   x1h  = (f16*)take(32768ull * 512 * 2);

  // weight packing (f32 (N,K) -> f16 (K,N)) + qkv bias concat
  packw_kernel<<<(1536 * 512 + 255) / 256, 256, 0, stream>>>(qkv_w, wqkv, 1536, 512);
  packw_kernel<<<(512 * 512 + 255) / 256, 256, 0, stream>>>(proj_w, wprj, 512, 512);
  packw_kernel<<<(2048 * 512 + 255) / 256, 256, 0, stream>>>(fc1_w, wfc1, 2048, 512);
  packw_kernel<<<(512 * 2048 + 255) / 256, 256, 0, stream>>>(fc2_w, wfc2, 512, 2048);
  build_qkv_bias<<<6, 256, 0, stream>>>(q_bias, v_bias, qkvb);

  // shift + window partition, QKV projection
  winpart_kernel<<<65536, 256, 0, stream>>>(x, xw);
  gemm_f16<<<dim3(12, 256), 256, 0, stream>>>(xw, wqkv, qkvb, qkvh, 32768, 1536, 512, 0);

  // continuous relative position bias
  cpb_table_kernel<<<961, 512, 0, stream>>>(cpb_w1, cpb_b1, cpb_w2, tbl);
  rpb_scatter_kernel<<<256, 256, 0, stream>>>(tbl, rpb);

  // cosine windowed attention (one WG per window x head), reuse xw for output
  attn_kernel<<<dim3(128, 16), 256, 0, stream>>>(qkvh, ls, rpb, xw);

  // proj, LN+residual (with inverse roll remap), MLP, final LN+residual
  gemm_f16<<<dim3(4, 256), 256, 0, stream>>>(xw, wprj, proj_b, prj, 32768, 512, 512, 2);
  ln_residual<<<32768, 256, 0, stream>>>(x, prj, n1g, n1b, x1, x1h, 1);
  gemm_f16<<<dim3(16, 256), 256, 0, stream>>>(x1h, wfc1, fc1_b, qkvh, 32768, 2048, 512, 1);
  gemm_f16<<<dim3(4, 256), 256, 0, stream>>>(qkvh, wfc2, fc2_b, prj, 32768, 512, 2048, 2);
  ln_residual<<<32768, 256, 0, stream>>>(x1, prj, n2g, n2b, out, nullptr, 0);
}
